// GPUOptimizedAttention_29231547417195
// MI455X (gfx1250) — matrix operand order repaired and checked by reading, not compiled, not run on hardware
//
#include <hip/hip_runtime.h>
#include <hip/hip_bf16.h>
#include <math.h>

typedef __attribute__((ext_vector_type(16))) _Float16 v16h;
typedef __attribute__((ext_vector_type(8)))  _Float16 v8h;
typedef __attribute__((ext_vector_type(8)))  float    v8f;

#define D_MODEL 1024
#define NHEAD   16
#define DK      64
#define SEQ     2048
#define BATCH   2
#define NTOK    (BATCH * SEQ)   // 4096

// padded LDS row strides (halves): chosen so the 16 active rows of a fragment
// read map to distinct bank groups (bank step 36 resp. 20, distinct mod 64
// for row deltas < 16) -> conflict-free at the 2-cycle wave32 floor.
#define KS_STRIDE 72   // 32 keys x 64 dims tile, 144 B rows
#define VS_STRIDE 40   // 64 dims x 32 keys tile, 80 B rows
#define AS_STRIDE 40   // 64 rows x 32 k tile,    80 B rows

// ---------------- WMMA helpers ----------------

__device__ __forceinline__ v8f wmma16x16x32(v16h a, v16h b, v8f c) {
  // D = A(16x32 f16) * B(32x16 f16) + C(16x16 f32)
  return __builtin_amdgcn_wmma_f32_16x16x32_f16(
      /*neg_a=*/false, a, /*neg_b=*/false, b,
      /*c_mod=*/(short)0, c, /*reuse_a=*/false, /*reuse_b=*/false);
}

__device__ __forceinline__ v16h join8(v8h lo, v8h hi) {
  return __builtin_shufflevector(lo, hi, 0,1,2,3,4,5,6,7,8,9,10,11,12,13,14,15);
}

// A-fragment (16x32, 16-bit) from a row pointer (global or LDS).
// Layout: lane<16 holds K{0..15}; lane>=16 holds K{16..31} (contiguous, matches load_b16).
__device__ __forceinline__ v16h load_a(const _Float16* rowptr, int k, int hf) {
  v8h lo = *(const v8h*)(rowptr + k + hf * 16);
  v8h hi = *(const v8h*)(rowptr + k + hf * 16 + 8);
  return join8(lo, hi);
}

// B-fragment: 16 contiguous halves (two 16-B pieces, only 16-B alignment needed).
__device__ __forceinline__ v16h load_b16(const _Float16* p) {
  v8h lo = *(const v8h*)(p);
  v8h hi = *(const v8h*)(p + 8);
  return join8(lo, hi);
}

// ---------------- prep kernels (f32 -> f16, transposes) ----------------

__global__ void k_cvt_f16(const float* __restrict__ src, _Float16* __restrict__ dst, int n) {
  int i = blockIdx.x * blockDim.x + threadIdx.x;
  if (i < n) dst[i] = (_Float16)src[i];
}

// src [rows][cols] f32  ->  dst [cols][rows] f16   (coalesced writes)
__global__ void k_transpose_f16(const float* __restrict__ src, _Float16* __restrict__ dst,
                                int rows, int cols) {
  int i = blockIdx.x * blockDim.x + threadIdx.x;
  if (i < rows * cols) {
    int c = i / rows;
    int r = i - c * rows;
    dst[(size_t)c * rows + r] = (_Float16)src[(size_t)r * cols + c];
  }
}

// ---------------- QKV projection GEMM ----------------
// 4-wave workgroup computes a 64x128 tile: the 64-row A panel is staged in LDS
// once per 32-k chunk and shared by all 4 waves (A L2 traffic / 4); each wave
// owns a 64x32 sub-tile (4x2 WMMA register blocking, 8 accumulators).

__global__ void __launch_bounds__(128, 1)
k_gemm_qkv(const _Float16* __restrict__ Xh,
           const _Float16* __restrict__ Wt,
           _Float16* __restrict__ Qh,
           _Float16* __restrict__ Kh,
           _Float16* __restrict__ Vt) {
  __shared__ _Float16 As[64 * AS_STRIDE];
  const int NPAN = (3 * D_MODEL) / 128;  // 24 column panels
  int tile = blockIdx.x;
  int m0 = (tile / NPAN) * 64;
  int tid = threadIdx.x;
  int wv = tid >> 5, lane = tid & 31;
  int l16 = lane & 15, hf = lane >> 4;
  int n0 = (tile % NPAN) * 128 + wv * 32;

  const _Float16* bcol = Wt + (size_t)(n0 + l16) * D_MODEL;  // +ns*16*D_MODEL folds to imm

  v8f acc[4][2] = {};
  for (int k = 0; k < D_MODEL; k += 32) {
    __syncthreads();  // previous chunk's LDS reads done
    {   // cooperative stage: 128 threads x 32 B = 64 rows x 32 halves
      int row = tid >> 1, seg = tid & 1;
      const _Float16* src = Xh + (size_t)(m0 + row) * D_MODEL + k + seg * 16;
      *(v8h*)(As + row * AS_STRIDE + seg * 16)     = *(const v8h*)(src);
      *(v8h*)(As + row * AS_STRIDE + seg * 16 + 8) = *(const v8h*)(src + 8);
    }
    __syncthreads();  // tile visible

    v16h a[4], b[2];
#pragma unroll
    for (int i = 0; i < 4; ++i)
      a[i] = load_a(As + (i * 16 + l16) * AS_STRIDE, 0, hf);
#pragma unroll
    for (int i = 0; i < 2; ++i)
      b[i] = load_b16(bcol + (size_t)i * 16 * D_MODEL + k + hf * 16);
#pragma unroll
    for (int ms = 0; ms < 4; ++ms)
#pragma unroll
      for (int ns = 0; ns < 2; ++ns)
        acc[ms][ns] = wmma16x16x32(a[ms], b[ns], acc[ms][ns]);
  }

#pragma unroll
  for (int ms = 0; ms < 4; ++ms) {
#pragma unroll
    for (int ns = 0; ns < 2; ++ns) {
      int n = n0 + ns * 16 + l16;      // qkv output channel 0..3071
      int which = n >> 10;             // 0=Q 1=K 2=V
      int rem = n & 1023;
      int h = rem >> 6, d = rem & 63;
#pragma unroll
      for (int r = 0; r < 8; ++r) {
        int mrow = m0 + ms * 16 + hf * 8 + r;    // token 0..4095
        int b = mrow >> 11, nn = mrow & 2047;
        float v = acc[ms][ns][r];
        size_t bh = (size_t)(b * NHEAD + h);
        if (which == 0)
          Qh[(bh * SEQ + nn) * DK + d] = (_Float16)(v * 0.125f);  // fold 1/sqrt(64)
        else if (which == 1)
          Kh[(bh * SEQ + nn) * DK + d] = (_Float16)v;
        else
          Vt[(bh * DK + d) * SEQ + nn] = (_Float16)v;             // V transposed
      }
    }
  }
}

// ---------------- attention core ----------------
// 4-wave workgroup covers 128 queries of one (b,h); each wave owns 32 queries
// (two 16-query tiles). K tiles (and V tiles in pass 2) are staged in LDS once
// per 32-key chunk and shared by all 4 waves -> K/V L2 traffic / 4.
// Computes S^T = K * Q^T so P = exp(S) lands directly in the 16-bit A-fragment
// layout required by the P @ V WMMA — no cross-lane transpose anywhere.

__global__ void __launch_bounds__(128, 1)
k_attention(const _Float16* __restrict__ Qh,
            const _Float16* __restrict__ Kh,
            const _Float16* __restrict__ Vt,
            _Float16* __restrict__ Oh) {
  __shared__ _Float16 Ks[32 * KS_STRIDE];  // 32 keys x 64 dims
  __shared__ _Float16 Vs[64 * VS_STRIDE];  // 64 dims x 32 keys

  int wid = blockIdx.x;             // bh*(SEQ/128) + panel
  int panel = wid & (SEQ / 128 - 1);
  int bh = wid >> 4;                // b*NHEAD + h
  int tid = threadIdx.x;
  int wv = tid >> 5, lane = tid & 31;
  int l16 = lane & 15, hf = lane >> 4;
  int q0 = panel * 128 + wv * 32;

  const _Float16* Qp = Qh + (size_t)bh * SEQ * DK;
  const _Float16* Kp = Kh + (size_t)bh * SEQ * DK;
  const _Float16* Vp = Vt + (size_t)bh * DK * SEQ;

  // B-fragments of Q^T (dims x 16 queries) for this wave's two query tiles.
  const _Float16* qrow = Qp + (size_t)(q0 + l16) * DK;
  v16h bq[2][2];
#pragma unroll
  for (int t = 0; t < 2; ++t) {
    bq[t][0] = load_b16(qrow + (size_t)t * 16 * DK + 0 + hf * 16);
    bq[t][1] = load_b16(qrow + (size_t)t * 16 * DK + 32 + hf * 16);
  }

  // ---- pass 1: per-query running max m and sum l (lane-local; q = l16) ----
  float m[2] = {-1e30f, -1e30f};
  float l[2] = {0.f, 0.f};
  for (int j = 0; j < SEQ; j += 32) {
    __syncthreads();
    {   // stage K tile: 128 threads x 32 B = 32 keys x 64 halves
      int key = tid >> 2, seg = tid & 3;
      const _Float16* src = Kp + (size_t)(j + key) * DK + seg * 16;
      *(v8h*)(Ks + key * KS_STRIDE + seg * 16)     = *(const v8h*)(src);
      *(v8h*)(Ks + key * KS_STRIDE + seg * 16 + 8) = *(const v8h*)(src + 8);
    }
    __syncthreads();

    v16h ak00 = load_a(Ks + l16 * KS_STRIDE, 0, hf);
    v16h ak01 = load_a(Ks + l16 * KS_STRIDE, 32, hf);
    v16h ak10 = load_a(Ks + (16 + l16) * KS_STRIDE, 0, hf);
    v16h ak11 = load_a(Ks + (16 + l16) * KS_STRIDE, 32, hf);
#pragma unroll
    for (int t = 0; t < 2; ++t) {
      v8f s0 = {};
      s0 = wmma16x16x32(ak00, bq[t][0], s0);
      s0 = wmma16x16x32(ak01, bq[t][1], s0);
      v8f s1 = {};
      s1 = wmma16x16x32(ak10, bq[t][0], s1);
      s1 = wmma16x16x32(ak11, bq[t][1], s1);
      float tm = s0[0];
#pragma unroll
      for (int r = 1; r < 8; ++r) tm = fmaxf(tm, s0[r]);
#pragma unroll
      for (int r = 0; r < 8; ++r) tm = fmaxf(tm, s1[r]);
      float mn = fmaxf(m[t], tm);
      float sum = 0.f;
#pragma unroll
      for (int r = 0; r < 8; ++r) sum += __expf(s0[r] - mn) + __expf(s1[r] - mn);
      l[t] = l[t] * __expf(m[t] - mn) + sum;
      m[t] = mn;
    }
  }
  // merge the two wave halves (each half saw a disjoint key subset)
  float mf[2], lf[2];
#pragma unroll
  for (int t = 0; t < 2; ++t) {
    float mo = __shfl_xor(m[t], 16, 32);
    float lo = __shfl_xor(l[t], 16, 32);
    mf[t] = fmaxf(m[t], mo);
    lf[t] = l[t] * __expf(m[t] - mf[t]) + lo * __expf(mo - mf[t]);
  }

  // ---- pass 2: out = softmax(S) @ V, 32 keys per iteration ----
  v8f acc[2][4] = {};
  for (int j = 0; j < SEQ; j += 32) {
    __syncthreads();
    {   // stage K tile
      int key = tid >> 2, seg = tid & 3;
      const _Float16* src = Kp + (size_t)(j + key) * DK + seg * 16;
      *(v8h*)(Ks + key * KS_STRIDE + seg * 16)     = *(const v8h*)(src);
      *(v8h*)(Ks + key * KS_STRIDE + seg * 16 + 8) = *(const v8h*)(src + 8);
    }
    {   // stage V tile: 128 threads x 32 B = 64 dims x 32 halves
      int d = tid >> 1, seg = tid & 1;
      const _Float16* src = Vp + (size_t)d * SEQ + j + seg * 16;
      *(v8h*)(Vs + d * VS_STRIDE + seg * 16)     = *(const v8h*)(src);
      *(v8h*)(Vs + d * VS_STRIDE + seg * 16 + 8) = *(const v8h*)(src + 8);
    }
    __syncthreads();

    v16h ak00 = load_a(Ks + l16 * KS_STRIDE, 0, hf);
    v16h ak01 = load_a(Ks + l16 * KS_STRIDE, 32, hf);
    v16h ak10 = load_a(Ks + (16 + l16) * KS_STRIDE, 0, hf);
    v16h ak11 = load_a(Ks + (16 + l16) * KS_STRIDE, 32, hf);

    // exp() and pack: S^T accumulator layout == A-fragment layout for P (16q x 32k)
    v16h pa[2];
#pragma unroll
    for (int t = 0; t < 2; ++t) {
      v8f s0 = {};
      s0 = wmma16x16x32(ak00, bq[t][0], s0);
      s0 = wmma16x16x32(ak01, bq[t][1], s0);
      v8f s1 = {};
      s1 = wmma16x16x32(ak10, bq[t][0], s1);
      s1 = wmma16x16x32(ak11, bq[t][1], s1);
#pragma unroll
      for (int r = 0; r < 8; ++r) {
        pa[t][r]     = (_Float16)__expf(s0[r] - mf[t]);
        pa[t][r + 8] = (_Float16)__expf(s1[r] - mf[t]);
      }
    }

#pragma unroll
    for (int dblk = 0; dblk < 4; ++dblk) {
      // B = V chunk (32 keys x 16 dims) from LDS; feeds both query tiles.
      v16h bv = join8(*(const v8h*)(Vs + (dblk * 16 + l16) * VS_STRIDE + hf * 8), *(const v8h*)(Vs + (dblk * 16 + l16) * VS_STRIDE + 16 + hf * 8));
      acc[0][dblk] = wmma16x16x32(pa[0], bv, acc[0][dblk]);
      acc[1][dblk] = wmma16x16x32(pa[1], bv, acc[1][dblk]);
    }
  }

  // normalize by 1/l (l for query q lives in lane q and q+16) and store f16
  int b = bh >> 4, h = bh & 15;
#pragma unroll
  for (int t = 0; t < 2; ++t) {
#pragma unroll
    for (int r = 0; r < 8; ++r) {
      int q = hf * 8 + r;
      float linv = 1.f / __shfl(lf[t], q, 32);
      size_t row = (size_t)(b * SEQ + q0 + t * 16 + q);
#pragma unroll
      for (int dblk = 0; dblk < 4; ++dblk) {
        Oh[row * D_MODEL + h * DK + dblk * 16 + l16] = (_Float16)(acc[t][dblk][r] * linv);
      }
    }
  }
}

// ---------------- output projection GEMM (+bias, f32 out) ----------------
// Same 4-wave / LDS-shared-A structure, 64x128 tile per workgroup.

__global__ void __launch_bounds__(128, 1)
k_gemm_out(const _Float16* __restrict__ Oh,
           const _Float16* __restrict__ Wt,   // Wout^T [1024][1024]
           const float* __restrict__ bias,
           float* __restrict__ out) {
  __shared__ _Float16 As[64 * AS_STRIDE];
  const int NPAN = D_MODEL / 128;  // 8 column panels
  int tile = blockIdx.x;
  int m0 = (tile / NPAN) * 64;
  int tid = threadIdx.x;
  int wv = tid >> 5, lane = tid & 31;
  int l16 = lane & 15, hf = lane >> 4;
  int n0 = (tile % NPAN) * 128 + wv * 32;

  const _Float16* bcol = Wt + (size_t)(n0 + l16) * D_MODEL;

  v8f acc[4][2] = {};
  for (int k = 0; k < D_MODEL; k += 32) {
    __syncthreads();
    {
      int row = tid >> 1, seg = tid & 1;
      const _Float16* src = Oh + (size_t)(m0 + row) * D_MODEL + k + seg * 16;
      *(v8h*)(As + row * AS_STRIDE + seg * 16)     = *(const v8h*)(src);
      *(v8h*)(As + row * AS_STRIDE + seg * 16 + 8) = *(const v8h*)(src + 8);
    }
    __syncthreads();

    v16h a[4], b[2];
#pragma unroll
    for (int i = 0; i < 4; ++i)
      a[i] = load_a(As + (i * 16 + l16) * AS_STRIDE, 0, hf);
#pragma unroll
    for (int i = 0; i < 2; ++i)
      b[i] = load_b16(bcol + (size_t)i * 16 * D_MODEL + k + hf * 16);
#pragma unroll
    for (int ms = 0; ms < 4; ++ms)
#pragma unroll
      for (int ns = 0; ns < 2; ++ns)
        acc[ms][ns] = wmma16x16x32(a[ms], b[ns], acc[ms][ns]);
  }

#pragma unroll
  for (int ms = 0; ms < 4; ++ms) {
#pragma unroll
    for (int ns = 0; ns < 2; ++ns) {
      int n = n0 + ns * 16 + l16;
      float bb = bias[n];
#pragma unroll
      for (int r = 0; r < 8; ++r) {
        int mrow = m0 + ms * 16 + hf * 8 + r;
        out[(size_t)mrow * D_MODEL + n] = acc[ms][ns][r] + bb;
      }
    }
  }
}

// ---------------- launcher ----------------

extern "C" void kernel_launch(void* const* d_in, const int* in_sizes, int n_in,
                              void* d_out, int out_size, void* d_ws, size_t ws_size,
                              hipStream_t stream) {
  (void)in_sizes; (void)n_in; (void)out_size; (void)ws_size;

  const float* x    = (const float*)d_in[0];   // [2,2048,1024]
  const float* Wqkv = (const float*)d_in[1];   // [1024,3072]
  const float* Wout = (const float*)d_in[2];   // [1024,1024]
  const float* bout = (const float*)d_in[3];   // [1024]
  float* out = (float*)d_out;

  char* ws = (char*)d_ws;
  size_t off = 0;
  _Float16* Xh    = (_Float16*)(ws + off); off += (size_t)NTOK * D_MODEL * 2;        // 8 MiB
  _Float16* WqkvT = (_Float16*)(ws + off); off += (size_t)3 * D_MODEL * D_MODEL * 2; // 6 MiB
  _Float16* WoutT = (_Float16*)(ws + off); off += (size_t)D_MODEL * D_MODEL * 2;     // 2 MiB
  _Float16* Qh    = (_Float16*)(ws + off); off += (size_t)NTOK * D_MODEL * 2;        // 8 MiB
  _Float16* Kh    = (_Float16*)(ws + off); off += (size_t)NTOK * D_MODEL * 2;        // 8 MiB
  _Float16* Vt    = (_Float16*)(ws + off); off += (size_t)NTOK * D_MODEL * 2;        // 8 MiB
  _Float16* Oh    = (_Float16*)(ws + off); off += (size_t)NTOK * D_MODEL * 2;        // 8 MiB
  // total 48 MiB

  int nX = NTOK * D_MODEL;
  k_cvt_f16<<<(nX + 255) / 256, 256, 0, stream>>>(x, Xh, nX);

  int nWq = D_MODEL * 3 * D_MODEL;
  k_transpose_f16<<<(nWq + 255) / 256, 256, 0, stream>>>(Wqkv, WqkvT, D_MODEL, 3 * D_MODEL);

  int nWo = D_MODEL * D_MODEL;
  k_transpose_f16<<<(nWo + 255) / 256, 256, 0, stream>>>(Wout, WoutT, D_MODEL, D_MODEL);

  k_gemm_qkv<<<(NTOK / 64) * ((3 * D_MODEL) / 128), 128, 0, stream>>>(Xh, WqkvT, Qh, Kh, Vt);

  k_attention<<<BATCH * NHEAD * (SEQ / 128), 128, 0, stream>>>(Qh, Kh, Vt, Oh);

  k_gemm_out<<<(NTOK / 64) * (D_MODEL / 128), 128, 0, stream>>>(Oh, WoutT, bout, out);
}
